// HGT_75943611728725
// MI455X (gfx1250) — matrix operand order repaired and checked by reading, not compiled, not run on hardware
//
#include <hip/hip_runtime.h>
#include <math.h>

// ---------------------------------------------------------------------------
// Types for gfx1250 WMMA (wave32): v16bf A/B fragments, v8f accumulator.
// ---------------------------------------------------------------------------
typedef __attribute__((ext_vector_type(16))) __bf16 v16bf;
typedef __attribute__((ext_vector_type(8)))  float  v8f;

__device__ __forceinline__ unsigned short f32_to_bf16(float f) {
  unsigned u = __float_as_uint(f);
  unsigned r = 0x7FFFu + ((u >> 16) & 1u);   // round to nearest even
  return (unsigned short)((u + r) >> 16);
}

union Frag { v16bf v; uint4 q[2]; unsigned short s[16]; };

// ---------------------------------------------------------------------------
// Tiled GEMM: C[M,N] = act(A[M,K]_bf16 * Bt[N,K]_bf16^T + bias) (+ skip mix)
// K % 32 == 0 (K is 256 or 768 here).
// Block = 128x128, 8 waves (4 along M, 2 along N), wave = 32x64 = 2x4 WMMA.
// Out-of-range rows/cols are CLAMPED (not masked): a clamped A-row only
// pollutes C-rows >= M and a clamped B-col only pollutes C-cols >= N, which
// are never stored. This keeps the K-loop free of exec-mask writes/branches.
// act: 0 = none, 1 = relu. If resid != null: out = s*out + (1-s)*resid,
// s = sigmoid(*skipP); resid has row stride N (only used with N==256).
// If Cbf != null, also store bf16 copy of the output (feeds next GEMM).
// ---------------------------------------------------------------------------
__global__ __launch_bounds__(256) void hgt_gemm_bf16(
    const unsigned short* __restrict__ A, int M, int K,
    const unsigned short* __restrict__ Bt, int N,
    const float* __restrict__ bias, float* __restrict__ C,
    unsigned short* __restrict__ Cbf,
    int act, const float* __restrict__ resid, const float* __restrict__ skipP)
{
  const int tid   = threadIdx.x;
  const int wave  = tid >> 5;
  const int lane  = tid & 31;
  const int wm    = wave & 3;          // 0..3 along M
  const int wn    = wave >> 2;         // 0..1 along N
  const int l15   = lane & 15;
  const int lhalf = lane >> 4;
  const int m0    = blockIdx.x * 128 + wm * 32;
  const int n0    = blockIdx.y * 128 + wn * 64;

  // Hoisted, clamped fragment base pointers (loop-invariant).
  // A lane layout (16x32 bf16): elems 0..7 -> K = kk + lhalf*8 + i; 8..15 -> +16
  const unsigned short* ap[2];
#pragma unroll
  for (int i = 0; i < 2; ++i) {
    int row = m0 + i * 16 + l15; row = row < M ? row : M - 1;
    ap[i] = A + (size_t)row * K + lhalf * 8;
  }
  // B lane layout (32x16 bf16): elems 0..7 -> K = kk + lhalf*8 + i; 8..15 -> +16, col n = l15
  const unsigned short* bp[4];
#pragma unroll
  for (int j = 0; j < 4; ++j) {
    int col = n0 + j * 16 + l15; col = col < N ? col : N - 1;
    bp[j] = Bt + (size_t)col * K + lhalf * 8;
  }

  v8f acc[2][4] = {};

  for (int kk = 0; kk < K; kk += 32) {
    Frag a[2], b[4];
#pragma unroll
    for (int i = 0; i < 2; ++i) {
      a[i].q[0] = *(const uint4*)(ap[i] + kk);
      a[i].q[1] = *(const uint4*)(ap[i] + kk + 16);
    }
#pragma unroll
    for (int j = 0; j < 4; ++j) {
      b[j].q[0] = *(const uint4*)(bp[j] + kk);
      b[j].q[1] = *(const uint4*)(bp[j] + kk + 16);
    }
#pragma unroll
    for (int i = 0; i < 2; ++i)
#pragma unroll
      for (int j = 0; j < 4; ++j)
        acc[i][j] = __builtin_amdgcn_wmma_f32_16x16x32_bf16(
            false, a[i].v, false, b[j].v, (short)0, acc[i][j], false, false);
  }

  float sk = 1.0f, skc = 0.0f;
  if (skipP) { sk = 1.0f / (1.0f + expf(-*skipP)); skc = 1.0f - sk; }

#pragma unroll
  for (int i = 0; i < 2; ++i) {
#pragma unroll
    for (int j = 0; j < 4; ++j) {
      int n = n0 + j * 16 + l15;
      if (n >= N) continue;
      float bv = bias ? bias[n] : 0.0f;
#pragma unroll
      for (int v = 0; v < 8; ++v) {
        int m = m0 + i * 16 + 8 * lhalf + v;
        if (m >= M) continue;
        float o = acc[i][j][v] + bv;
        if (act == 1) o = fmaxf(o, 0.0f);
        if (resid) o = sk * o + skc * resid[(size_t)m * N + n];
        C[(size_t)m * N + n] = o;
        if (Cbf) Cbf[(size_t)m * N + n] = f32_to_bf16(o);
      }
    }
  }
}

// ---------------------------------------------------------------------------
// Weight prep: transpose + f32->bf16.  Wt[n*K + k] = bf16(W[k*N + n])
// ---------------------------------------------------------------------------
__global__ void hgt_wconv(const float* __restrict__ W, int K, int N,
                          unsigned short* __restrict__ Wt) {
  long long t = (long long)blockIdx.x * blockDim.x + threadIdx.x;
  if (t >= (long long)K * N) return;
  int k = (int)(t / N), n = (int)(t % N);
  Wt[(size_t)n * K + k] = f32_to_bf16(W[t]);
}

// Combined rel-weights: Wc = Wlin(256x256) @ blockdiag(A[8][32][32]),
// stored transposed bf16 [n][i]; bias bc[n] = sum_d b[32h+d]*A[h][d][n%32].
__global__ void hgt_wcombine(const float* __restrict__ W, const float* __restrict__ b,
                             const float* __restrict__ A,
                             unsigned short* __restrict__ Wct, float* __restrict__ bc) {
  int t = blockIdx.x * blockDim.x + threadIdx.x;
  if (t >= 256 * 256) return;
  int n = t >> 8, i = t & 255;
  int h = n >> 5, c = n & 31;
  const float* Ah = A + h * 1024;          // [d][c], stride 32
  const float* Wr = W + (size_t)i * 256 + h * 32;
  float s = 0.0f;
#pragma unroll 8
  for (int d = 0; d < 32; ++d) s += Wr[d] * Ah[d * 32 + c];
  Wct[(size_t)n * 256 + i] = f32_to_bf16(s);
  if (i == 0) {
    float sb = 0.0f;
    for (int d = 0; d < 32; ++d) sb += b[h * 32 + d] * Ah[d * 32 + c];
    bc[n] = sb;
  }
}

// ---------------------------------------------------------------------------
// Elementwise helpers
// ---------------------------------------------------------------------------
__global__ void hgt_fill(float* __restrict__ p, long long n, float v) {
  long long t = (long long)blockIdx.x * blockDim.x + threadIdx.x;
  if (t < n) p[t] = v;
}

__global__ void hgt_cvt(const float* __restrict__ in,
                        unsigned short* __restrict__ out, long long n) {
  long long t = (long long)blockIdx.x * blockDim.x + threadIdx.x;
  if (t < n) out[t] = f32_to_bf16(in[t]);
}

// exact gelu, emits bf16 (only consumed as a GEMM A-operand)
__global__ void hgt_gelu(const float* __restrict__ in,
                         unsigned short* __restrict__ out, long long n) {
  long long t = (long long)blockIdx.x * blockDim.x + threadIdx.x;
  if (t < n) {
    float x = in[t];
    out[t] = f32_to_bf16(0.5f * x * (1.0f + erff(x * 0.70710678118654752f)));
  }
}

// ---------------------------------------------------------------------------
// Edge passes (segment softmax per edge type over dst nodes)
// ---------------------------------------------------------------------------
__device__ __forceinline__ void atomicMaxF(float* addr, float val) {
  if (val >= 0.0f) atomicMax((int*)addr, __float_as_int(val));
  else             atomicMin((unsigned int*)addr, __float_as_uint(val));
}

// wave per edge: per-head dot(q[dst], krel[src]) * p_rel / sqrt(32);
// store logit, atomic-max into mb[dst*8+h].
__global__ void hgt_edge_logits(const int* __restrict__ ei, int E,
                                const float* __restrict__ q,
                                const float* __restrict__ krel,
                                const float* __restrict__ prel,
                                float* __restrict__ lg, float* __restrict__ mb) {
  long long gt = (long long)blockIdx.x * blockDim.x + threadIdx.x;
  int e = (int)(gt >> 5);
  int lane = threadIdx.x & 31;
  if (e >= E) return;
  int si = ei[e], di = ei[E + e];
  const float4* qp = (const float4*)(q + (size_t)di * 256) + lane * 2;
  const float4* kp = (const float4*)(krel + (size_t)si * 256) + lane * 2;
  float4 q0 = qp[0], q1 = qp[1], k0 = kp[0], k1 = kp[1];
  float s = q0.x * k0.x + q0.y * k0.y + q0.z * k0.z + q0.w * k0.w
          + q1.x * k1.x + q1.y * k1.y + q1.z * k1.z + q1.w * k1.w;
  s += __shfl_xor(s, 1, 32);
  s += __shfl_xor(s, 2, 32);                 // lanes 4h..4h+3 hold head-h dot
  int h = lane >> 2;
  if ((lane & 3) == 0) {
    float v = s * prel[h] * 0.17677669529663687f;  // 1/sqrt(32)
    lg[(size_t)e * 8 + h] = v;
    atomicMaxF(mb + (size_t)di * 8 + h, v);
  }
}

// thread per (edge, head): ex = exp(logit - max); atomic-add into denom.
__global__ void hgt_edge_exp(const int* __restrict__ ei, int E,
                             float* __restrict__ lg, const float* __restrict__ mb,
                             float* __restrict__ db) {
  long long t = (long long)blockIdx.x * blockDim.x + threadIdx.x;
  if (t >= (long long)E * 8) return;
  int e = (int)(t >> 3), h = (int)(t & 7);
  int di = ei[E + e];
  float ex = expf(lg[t] - mb[(size_t)di * 8 + h]);
  lg[t] = ex;
  atomicAdd(db + (size_t)di * 8 + h, ex);
}

// wave per edge: agg[dst] += vrel[src] * (ex / denom), 8 floats per lane.
__global__ void hgt_edge_agg(const int* __restrict__ ei, int E,
                             const float* __restrict__ lg, const float* __restrict__ db,
                             const float* __restrict__ vrel, float* __restrict__ agg) {
  long long gt = (long long)blockIdx.x * blockDim.x + threadIdx.x;
  int e = (int)(gt >> 5);
  int lane = threadIdx.x & 31;
  if (e >= E) return;
  int si = ei[e], di = ei[E + e];
  int h = lane >> 2;
  float w = lg[(size_t)e * 8 + h] / (db[(size_t)di * 8 + h] + 1e-16f);
  const float4* vp = (const float4*)(vrel + (size_t)si * 256) + lane * 2;
  float4 a0 = vp[0], a1 = vp[1];
  float* o = agg + (size_t)di * 256 + lane * 8;
  atomicAdd(o + 0, a0.x * w); atomicAdd(o + 1, a0.y * w);
  atomicAdd(o + 2, a0.z * w); atomicAdd(o + 3, a0.w * w);
  atomicAdd(o + 4, a1.x * w); atomicAdd(o + 5, a1.y * w);
  atomicAdd(o + 6, a1.z * w); atomicAdd(o + 7, a1.w * w);
}

// ---------------------------------------------------------------------------
// Host orchestration
// ---------------------------------------------------------------------------
extern "C" void kernel_launch(void* const* d_in, const int* in_sizes, int n_in,
                              void* d_out, int out_size, void* d_ws, size_t ws_size,
                              hipStream_t stream) {
  (void)n_in; (void)out_size; (void)ws_size;
  auto F = [&](int i) { return (const float*)d_in[i]; };

  // node types (my order): 0=hierarchy 1=protocol 2=impression 3=treatment
  const float* X[4]; int Nn[4];
  for (int nt = 0; nt < 4; ++nt) { X[nt] = F(nt); Nn[nt] = in_sizes[nt] / 768; }
  const int* EI[6]; int Ne[6];
  for (int j = 0; j < 6; ++j) { EI[j] = (const int*)d_in[4 + j]; Ne[j] = in_sizes[4 + j] / 2; }
  // edge types (my order = EDGE_TYPES order): p2h p2i p2t h2p i2p t2p
  const int esrc[6] = {1, 1, 1, 0, 2, 3};
  const int edst[6] = {0, 2, 3, 1, 1, 1};
  // params flattened in JAX pytree order (dict keys sorted alphabetically)
  const int s2nt[4] = {0, 2, 1, 3};          // sorted nt -> my nt
  const int s2et[6] = {3, 4, 1, 0, 2, 5};    // sorted et-key -> my et

  struct Lp {
    const float *a_b[4], *a_w[4], *a_rel[6], *k_b[4], *k_w[4], *m_rel[6],
                *p_rel[6], *q_b[4], *q_w[4], *skip[4], *v_b[4], *v_w[4];
  } L[2];
  int idx = 10;
  for (int l = 0; l < 2; ++l) {
    for (int s = 0; s < 4; ++s) { int nt = s2nt[s]; L[l].a_b[nt] = F(idx++); L[l].a_w[nt] = F(idx++); }
    for (int s = 0; s < 6; ++s)   L[l].a_rel[s2et[s]] = F(idx++);
    for (int s = 0; s < 4; ++s) { int nt = s2nt[s]; L[l].k_b[nt] = F(idx++); L[l].k_w[nt] = F(idx++); }
    for (int s = 0; s < 6; ++s)   L[l].m_rel[s2et[s]] = F(idx++);
    for (int s = 0; s < 6; ++s)   L[l].p_rel[s2et[s]] = F(idx++);
    for (int s = 0; s < 4; ++s) { int nt = s2nt[s]; L[l].q_b[nt] = F(idx++); L[l].q_w[nt] = F(idx++); }
    for (int s = 0; s < 4; ++s) { int nt = s2nt[s]; L[l].skip[nt] = F(idx++); }
    for (int s = 0; s < 4; ++s) { int nt = s2nt[s]; L[l].v_b[nt] = F(idx++); L[l].v_w[nt] = F(idx++); }
  }
  const float *li_b[4], *li_w[4];
  for (int s = 0; s < 4; ++s) { int nt = s2nt[s]; li_b[nt] = F(idx++); li_w[nt] = F(idx++); }
  const float* lo_b = F(idx++);
  const float* lo_w = F(idx++);

  // ---- workspace carve (≈330 MB) ----
  char* wsb = (char*)d_ws;
  size_t off = 0;
  auto carve = [&](size_t bytes) -> char* {
    off = (off + 255) & ~(size_t)255;
    char* p = wsb + off; off += bytes; return p;
  };
  const int ntOff[4] = {0, 2048, 22144, 32256};  // 128-padded prefix sums
  const int NPtot = 37376;
  float* hA   = (float*)carve((size_t)NPtot * 256 * 4);
  float* hB   = (float*)carve((size_t)NPtot * 256 * 4);
  float* qg   = (float*)carve((size_t)NPtot * 256 * 4);
  float* agg  = (float*)carve((size_t)NPtot * 256 * 4);
  float* krel = (float*)carve((size_t)20096 * 256 * 4);
  float* vrel = (float*)carve((size_t)20096 * 256 * 4);
  float* exb  = (float*)carve((size_t)200000 * 8 * 4);
  float* mb   = (float*)carve((size_t)20096 * 8 * 4);
  float* db   = (float*)carve((size_t)20096 * 8 * 4);
  unsigned short* hAbf = (unsigned short*)carve((size_t)NPtot * 256 * 2);
  unsigned short* hBbf = (unsigned short*)carve((size_t)NPtot * 256 * 2);
  unsigned short* gbf  = (unsigned short*)carve((size_t)NPtot * 256 * 2);
  unsigned short* xbf[4];
  for (int nt = 0; nt < 4; ++nt)
    xbf[nt] = (unsigned short*)carve((size_t)Nn[nt] * 768 * 2);
  unsigned short* wt_li[4];
  for (int nt = 0; nt < 4; ++nt) wt_li[nt] = (unsigned short*)carve((size_t)768 * 256 * 2);
  unsigned short *wt_q[2][4], *wt_a[2][4], *wt_kc[2][6], *wt_vc[2][6];
  float *bc_k[2][6], *bc_v[2][6];
  for (int l = 0; l < 2; ++l) {
    for (int nt = 0; nt < 4; ++nt) {
      wt_q[l][nt] = (unsigned short*)carve((size_t)256 * 256 * 2);
      wt_a[l][nt] = (unsigned short*)carve((size_t)256 * 256 * 2);
    }
    for (int et = 0; et < 6; ++et) {
      wt_kc[l][et] = (unsigned short*)carve((size_t)256 * 256 * 2);
      wt_vc[l][et] = (unsigned short*)carve((size_t)256 * 256 * 2);
      bc_k[l][et] = (float*)carve(256 * 4);
      bc_v[l][et] = (float*)carve(256 * 4);
    }
  }
  unsigned short* wt_out = (unsigned short*)carve((size_t)2500 * 256 * 2);

  auto cdiv = [](long long a, long long b) { return (int)((a + b - 1) / b); };
  auto gemm = [&](const unsigned short* Ain, int M, int K,
                  const unsigned short* Bt, int N, const float* bias,
                  float* Cout, unsigned short* Cbf, int act,
                  const float* resid, const float* skipP) {
    dim3 g(cdiv(M, 128), cdiv(N, 128));
    hgt_gemm_bf16<<<g, 256, 0, stream>>>(Ain, M, K, Bt, N, bias, Cout, Cbf,
                                         act, resid, skipP);
  };

  // ---- weight prep (bf16 transpose + combined rel weights) ----
  for (int nt = 0; nt < 4; ++nt)
    hgt_wconv<<<cdiv(768 * 256, 256), 256, 0, stream>>>(li_w[nt], 768, 256, wt_li[nt]);
  for (int l = 0; l < 2; ++l)
    for (int nt = 0; nt < 4; ++nt) {
      hgt_wconv<<<cdiv(65536, 256), 256, 0, stream>>>(L[l].q_w[nt], 256, 256, wt_q[l][nt]);
      hgt_wconv<<<cdiv(65536, 256), 256, 0, stream>>>(L[l].a_w[nt], 256, 256, wt_a[l][nt]);
    }
  hgt_wconv<<<cdiv(256 * 2500, 256), 256, 0, stream>>>(lo_w, 256, 2500, wt_out);
  for (int l = 0; l < 2; ++l)
    for (int et = 0; et < 6; ++et) {
      int s = esrc[et];
      hgt_wcombine<<<256, 256, 0, stream>>>(L[l].k_w[s], L[l].k_b[s], L[l].a_rel[et],
                                            wt_kc[l][et], bc_k[l][et]);
      hgt_wcombine<<<256, 256, 0, stream>>>(L[l].v_w[s], L[l].v_b[s], L[l].m_rel[et],
                                            wt_vc[l][et], bc_v[l][et]);
    }

  // ---- input features -> bf16 (once), then lin_in + relu (f32 + bf16 out) --
  for (int nt = 0; nt < 4; ++nt)
    hgt_cvt<<<cdiv((long long)Nn[nt] * 768, 256), 256, 0, stream>>>(
        X[nt], xbf[nt], (long long)Nn[nt] * 768);
  for (int nt = 0; nt < 4; ++nt)
    gemm(xbf[nt], Nn[nt], 768, wt_li[nt], 256, li_b[nt],
         hA + (size_t)ntOff[nt] * 256, hAbf + (size_t)ntOff[nt] * 256,
         1, nullptr, nullptr);

  // ---- 2 HGT layers ----
  float* hcur = hA; float* hnxt = hB;
  unsigned short* hcurbf = hAbf; unsigned short* hnxtbf = hBbf;
  for (int l = 0; l < 2; ++l) {
    for (int nt = 0; nt < 4; ++nt)
      gemm(hcurbf + (size_t)ntOff[nt] * 256, Nn[nt], 256, wt_q[l][nt], 256,
           L[l].q_b[nt], qg + (size_t)ntOff[nt] * 256, nullptr, 0, nullptr, nullptr);
    hgt_fill<<<cdiv((long long)NPtot * 256, 256), 256, 0, stream>>>(
        agg, (long long)NPtot * 256, 0.0f);

    for (int et = 0; et < 6; ++et) {
      int s = esrc[et], d = edst[et], E = Ne[et];
      gemm(hcurbf + (size_t)ntOff[s] * 256, Nn[s], 256, wt_kc[l][et], 256,
           bc_k[l][et], krel, nullptr, 0, nullptr, nullptr);
      gemm(hcurbf + (size_t)ntOff[s] * 256, Nn[s], 256, wt_vc[l][et], 256,
           bc_v[l][et], vrel, nullptr, 0, nullptr, nullptr);
      hgt_fill<<<cdiv((long long)Nn[d] * 8, 256), 256, 0, stream>>>(
          mb, (long long)Nn[d] * 8, -INFINITY);
      hgt_fill<<<cdiv((long long)Nn[d] * 8, 256), 256, 0, stream>>>(
          db, (long long)Nn[d] * 8, 0.0f);
      hgt_edge_logits<<<cdiv((long long)E * 32, 256), 256, 0, stream>>>(
          EI[et], E, qg + (size_t)ntOff[d] * 256, krel, L[l].p_rel[et], exb, mb);
      hgt_edge_exp<<<cdiv((long long)E * 8, 256), 256, 0, stream>>>(
          EI[et], E, exb, mb, db);
      hgt_edge_agg<<<cdiv((long long)E * 32, 256), 256, 0, stream>>>(
          EI[et], E, exb, db, vrel, agg + (size_t)ntOff[d] * 256);
    }

    for (int nt = 0; nt < 4; ++nt) {
      hgt_gelu<<<cdiv((long long)Nn[nt] * 256, 256), 256, 0, stream>>>(
          agg + (size_t)ntOff[nt] * 256, gbf + (size_t)ntOff[nt] * 256,
          (long long)Nn[nt] * 256);
      gemm(gbf + (size_t)ntOff[nt] * 256, Nn[nt], 256, wt_a[l][nt], 256,
           L[l].a_b[nt], hnxt + (size_t)ntOff[nt] * 256,
           hnxtbf + (size_t)ntOff[nt] * 256, 0,
           hcur + (size_t)ntOff[nt] * 256, L[l].skip[nt]);
    }
    float* t = hcur; hcur = hnxt; hnxt = t;
    unsigned short* tb = hcurbf; hcurbf = hnxtbf; hnxtbf = tb;
  }

  // ---- lin_out on protocol (bf16 activations, 20 column blocks) ----
  gemm(hcurbf + (size_t)ntOff[1] * 256, Nn[1], 256, wt_out, 2500, lo_b,
       (float*)d_out, nullptr, 0, nullptr, nullptr);
}
